// ChangeDetector_73478300500673
// MI455X (gfx1250) — matrix operand order repaired and checked by reading, not compiled, not run on hardware
//
#include <hip/hip_runtime.h>
#include <hip/hip_bf16.h>

// ---------------------------------------------------------------------------
// MI455X (gfx1250): wave32, WMMA f16 16x16x32 for all GEMMs, async
// global->LDS staging (GLOBAL_LOAD_ASYNC_TO_LDS_B128, ASYNCcnt) with
// double-buffered k-slices, and TDM (TENSOR_LOAD_TO_LDS, TENSORcnt) for the
// attention K-tile DMA.
// ---------------------------------------------------------------------------

typedef __attribute__((ext_vector_type(16))) _Float16 v16h;
typedef __attribute__((ext_vector_type(8)))  _Float16 v8h;
typedef __attribute__((ext_vector_type(8)))  float    v8f;
typedef __attribute__((ext_vector_type(4)))  unsigned v4u;
typedef __attribute__((ext_vector_type(8)))  int      v8i;
typedef __attribute__((ext_vector_type(4)))  int      v4i;

#define NB   32          // batch
#define LL   196         // sequence (14*14)
#define DD   512         // model dim
#define CC   1024        // conv input channels
#define MM   (NB * LL)   // 6272 rows
#define DHD  64          // head dim
#define NHD  8           // heads

// ----------------------------- WMMA helpers --------------------------------

__device__ __forceinline__ v8f vzero() {
  v8f z;
#pragma unroll
  for (int j = 0; j < 8; ++j) z[j] = 0.0f;
  return z;
}

// A fragment 16x32 f16 (ISA 7.12.2): lane<16 -> M=lane, K in {0..7,16..23};
// lane>=16 -> M=lane-16, K in {8..15,24..31}. Two 16B LDS loads per lane.
__device__ __forceinline__ v16h frag_a(const _Float16* p, int stride) {
  const int lane = threadIdx.x & 31;
  const _Float16* q = p + (lane & 15) * stride + ((lane >> 4) << 3);
  v8h lo = *(const v8h*)q;
  v8h hi = *(const v8h*)(q + 16);
  v16h r;
#pragma unroll
  for (int i = 0; i < 8; ++i) { r[i] = lo[i]; r[i + 8] = hi[i]; }
  return r;
}

// Same A fragment but source data is f32 in LDS (softmax probabilities).
__device__ __forceinline__ v16h frag_a_f32(const float* p, int stride) {
  const int lane = threadIdx.x & 31;
  const float* q = p + (lane & 15) * stride + ((lane >> 4) << 3);
  v16h r;
#pragma unroll
  for (int i = 0; i < 8; ++i) { r[i] = (_Float16)q[i]; r[i + 8] = (_Float16)q[i + 16]; }
  return r;
}

// B fragment 32x16 f16 from B^T stored row-major [n][k]: lane<16 -> N=lane,
// K in {0..7,16..23}; lane>=16 -> N=lane-16, K in {8..15,24..31}. Two 16B loads.
__device__ __forceinline__ v16h frag_bt(const _Float16* p, int stride) {
  const int lane = threadIdx.x & 31;
  const _Float16* q = p + (lane & 15) * stride + ((lane >> 4) << 3);
  v8h lo = *(const v8h*)q;
  v8h hi = *(const v8h*)(q + 16);
  v16h r;
#pragma unroll
  for (int i = 0; i < 8; ++i) { r[i] = lo[i]; r[i + 8] = hi[i]; }
  return r;
}

__device__ __forceinline__ v8f wmma16(v16h a, v16h b, v8f c) {
  return __builtin_amdgcn_wmma_f32_16x16x32_f16(false, a, false, b, (short)0, c,
                                                false, false);
}

// C/D layout: lane<16 -> N=lane, M=j (VGPR j); lane>=16 -> N=lane-16, M=8+j.
__device__ __forceinline__ void cpos(int& n, int& mb) {
  const int lane = threadIdx.x & 31;
  n = lane & 15;
  mb = (lane >> 4) << 3;
}

// ------------------ async global -> LDS copy (CDNA5) -----------------------
// GLOBAL_LOAD_ASYNC_TO_LDS_B128: per-lane 16B, LDS dest addr in VDST VGPR,
// 64-bit global addr in VADDR pair, GV mode (saddr = off). Tracked by
// ASYNCcnt; completed by s_wait_asynccnt 0 before the workgroup barrier.
__device__ __forceinline__ void async_g2l_b128(void* lds, const void* g) {
  unsigned loff = (unsigned)(size_t)lds;                 // LDS byte offset
  unsigned long long ga = (unsigned long long)(size_t)g; // global address
  asm volatile("global_load_async_to_lds_b128 %0, %1, off"
               :: "v"(loff), "v"(ga) : "memory");
}
__device__ __forceinline__ void wait_async0() {
  asm volatile("s_wait_asynccnt 0" ::: "memory");
}

// ---------------- TDM: 2D f16 tile DMA global -> LDS (CDNA5) ---------------
// Builds a 2-group Tensor DMA Descriptor (CDNA5 ISA sec.8):
//  group0: count=1 | lds_addr | global_addr(57b) | type=2
//  group1: data_size=2B, tensor_dim0/1, tile_dim0/1, dim0 row stride.
// Issued once per workgroup (wave 0); completion via s_wait_tensorcnt 0,
// then the workgroup barrier publishes the LDS tile to all waves.
// This toolchain exposes the 6-arg builtin:
//   (uint32x4 g0, int32x8 g1, int32x4 g2, int32x4 g3, int32x8, i32 cpol)
#if defined(__gfx1250__) && __has_builtin(__builtin_amdgcn_tensor_load_to_lds)
#define HAVE_TDM 1
__device__ __forceinline__ void tdm_load_2d_f16(
    void* lds, const void* gaddr, unsigned tileW, unsigned tileH,
    unsigned rowStride, unsigned tensorW, unsigned tensorH) {
  unsigned long long ga = (unsigned long long)(size_t)gaddr;
  v4u g0;
  g0[0] = 1u;                                   // count=1, user descriptor
  g0[1] = (unsigned)(size_t)lds;                // lds_addr (bytes)
  g0[2] = (unsigned)(ga & 0xffffffffu);         // global_addr[31:0]
  g0[3] = (unsigned)((ga >> 32) & 0x01ffffffu)  // global_addr[56:32]
          | (2u << 30);                         // type=2 ("image")
  v8i g1;
  g1[0] = (int)(1u << 16);                      // data_size=1 -> 2 bytes
  g1[1] = (int)((tensorW & 0xffffu) << 16);     // tensor_dim0[15:0] @ bits 63:48
  g1[2] = (int)(((tensorW >> 16) & 0xffffu)     // tensor_dim0[31:16]
                | ((tensorH & 0xffffu) << 16)); // tensor_dim1[15:0]
  g1[3] = (int)(((tensorH >> 16) & 0xffffu)     // tensor_dim1[31:16]
                | ((tileW & 0xffffu) << 16));   // tile_dim0
  g1[4] = (int)(tileH & 0xffffu);               // tile_dim1 (tile_dim2 = 0)
  g1[5] = (int)rowStride;                       // tensor_dim0_stride[31:0]
  g1[6] = 0;                                    // stride[47:32], dim1_stride lo
  g1[7] = 0;                                    // dim1_stride hi (unused, 2D)
  v4i z4 = {0, 0, 0, 0};
  v8i z8 = {0, 0, 0, 0, 0, 0, 0, 0};
  __builtin_amdgcn_tensor_load_to_lds(g0, g1, z4, z4, z8, 0);
}
__device__ __forceinline__ void wait_tensor0() {
#if __has_builtin(__builtin_amdgcn_s_wait_tensorcnt)
  __builtin_amdgcn_s_wait_tensorcnt(0);
#else
  asm volatile("s_wait_tensorcnt 0" ::: "memory");
#endif
}
#else
#define HAVE_TDM 0
#endif

// Order-preserving float<->uint mapping for atomic max reductions in LDS.
__device__ __forceinline__ unsigned ordkey(float f) {
  unsigned u = __float_as_uint(f);
  return (u & 0x80000000u) ? ~u : (u | 0x80000000u);
}
__device__ __forceinline__ float orddec(unsigned u) {
  unsigned v = (u & 0x80000000u) ? (u & 0x7FFFFFFFu) : ~u;
  return __uint_as_float(v);
}

// -------------------- weight conversion (fp32 -> f16, transposed) ----------
// wT[(layer*3+which)*D*D + n*D + k] = W[layer][k][n]  (B^T layout for WMMA)
// fcT[n*1024 + k] = fc_w[k][n]
__global__ __launch_bounds__(256) void wconv_kernel(
    const float* Wq, const float* Wk, const float* Wv, const float* fcw,
    _Float16* wT, _Float16* fcT) {
  int i = blockIdx.x * 256 + threadIdx.x;
  const int awn = 2 * 3 * DD * DD;
  if (i < awn) {
    int k = i & (DD - 1);
    int n = (i >> 9) & (DD - 1);
    int lw = i >> 18;              // layer*3 + which
    int layer = lw / 3, which = lw - layer * 3;
    const float* W = (which == 0) ? Wq : (which == 1) ? Wk : Wv;
    wT[i] = (_Float16)W[((size_t)layer * DD + k) * DD + n];
  } else {
    int j = i - awn;
    if (j < DD * CC) {
      int k = j & (CC - 1);
      int n = j >> 10;
      fcT[j] = (_Float16)fcw[(size_t)k * DD + n];
    }
  }
}

// ------------------- embed: 1x1 conv GEMM + bias + pos-emb -----------------
// grid (98, 8, 2), 256 thr. BM=BN=64, K=1024. fp32 source needs VGPR
// conversion, so staging stays synchronous here.
__global__ __launch_bounds__(256) void embed_kernel(
    const float* in1, const float* in2, const float* convw, const float* convb,
    const float* wemb, const float* hemb, float* pre1, float* pre2,
    float* x1f, float* x2f, _Float16* x1h, _Float16* x2h) {
  __shared__ __align__(16) _Float16 As[64 * 32];
  __shared__ __align__(16) _Float16 Bs[64 * 32];
  const int mt = blockIdx.x, ntb = blockIdx.y, which = blockIdx.z;
  const float* in = which ? in2 : in1;
  float* pre = which ? pre2 : pre1;
  float* xf = which ? x2f : x1f;
  _Float16* xh = which ? x2h : x1h;
  const int tid = threadIdx.x, wave = tid >> 5;
  const int m0 = mt * 64, n0 = ntb * 64;

  int tm[2], tn[2];
  v8f acc[2];
#pragma unroll
  for (int i = 0; i < 2; ++i) {
    int t = wave * 2 + i;
    tm[i] = t >> 2; tn[i] = t & 3;
    acc[i] = vzero();
  }

  const int mloc = tid & 63, kseg = tid >> 6;  // kseg 0..3
  const int mg = m0 + mloc;
  const int bb = mg / LL, hw = mg - bb * LL;
  const float* arow = in + (size_t)bb * CC * LL + hw;  // + k*LL

  for (int k0 = 0; k0 < CC; k0 += 32) {
    __syncthreads();
#pragma unroll
    for (int i = 0; i < 8; ++i) {
      int k = kseg * 8 + i;
      As[mloc * 32 + k] = (_Float16)arow[(size_t)(k0 + k) * LL];
    }
    {
      const float* brow = convw + (size_t)(n0 + mloc) * CC + k0 + kseg * 8;
      __builtin_prefetch(brow + 32, 0, 1);
#pragma unroll
      for (int i = 0; i < 8; ++i)
        Bs[mloc * 32 + kseg * 8 + i] = (_Float16)brow[i];
    }
    __syncthreads();
#pragma unroll
    for (int i = 0; i < 2; ++i) {
      v16h a = frag_a(As + tm[i] * 16 * 32, 32);
      v16h b = frag_bt(Bs + tn[i] * 16 * 32, 32);
      acc[i] = wmma16(a, b, acc[i]);
    }
  }

  int n_, mb; cpos(n_, mb);
#pragma unroll
  for (int i = 0; i < 2; ++i) {
#pragma unroll
    for (int j = 0; j < 8; ++j) {
      int mm = m0 + tm[i] * 16 + mb + j;
      int oo = n0 + tn[i] * 16 + n_;
      int b2 = mm / LL, hw2 = mm - b2 * LL;
      int hy = hw2 / 14, hx = hw2 - hy * 14;
      float pe = (oo < 256) ? wemb[hx * 256 + oo] : hemb[hy * 256 + (oo - 256)];
      float val = acc[i][j] + convb[oo] + pe;
      size_t idx = (size_t)mm * DD + oo;
      pre[idx] = val; xf[idx] = val; xh[idx] = (_Float16)val;
    }
  }
}

// -------------------- QKV projection GEMM (f16 in, f16 out) ----------------
// grid (98, 8, 3): z selects Q/K/V. BM=BN=64, K=512.
// Async double-buffered staging: copy slice k+1 into LDS buf^1 while the
// WMMAs consume buf, then s_wait_asynccnt 0 + barrier.
__global__ __launch_bounds__(256) void qkv_kernel(
    const _Float16* xq, const _Float16* xkv, const _Float16* wT,
    const float* bq, const float* bk, const float* bv,
    _Float16* qh, _Float16* kh, _Float16* vh) {
  __shared__ __align__(16) _Float16 As[2][64 * 32];
  __shared__ __align__(16) _Float16 Bs[2][64 * 32];
  const int which = blockIdx.z;
  const _Float16* A = (which == 0) ? xq : xkv;
  const _Float16* BT = wT + (size_t)which * DD * DD;
  const float* bias = (which == 0) ? bq : (which == 1) ? bk : bv;
  _Float16* out = (which == 0) ? qh : (which == 1) ? kh : vh;
  const int tid = threadIdx.x, wave = tid >> 5;
  const int m0 = blockIdx.x * 64, n0 = blockIdx.y * 64;

  int tm[2], tn[2];
  v8f acc[2];
#pragma unroll
  for (int i = 0; i < 2; ++i) {
    int t = wave * 2 + i;
    tm[i] = t >> 2; tn[i] = t & 3;
    acc[i] = vzero();
  }
  const int row = tid >> 2, kb = (tid & 3) * 8;
  const _Float16* agp = A + (size_t)(m0 + row) * DD + kb;
  const _Float16* bgp = BT + (size_t)(n0 + row) * DD + kb;

  async_g2l_b128(&As[0][row * 32 + kb], agp);
  async_g2l_b128(&Bs[0][row * 32 + kb], bgp);
  wait_async0();
  __syncthreads();

  int buf = 0;
  for (int k0 = 0; k0 < DD; k0 += 32, buf ^= 1) {
    const int k1 = k0 + 32;
    if (k1 < DD) {
      async_g2l_b128(&As[buf ^ 1][row * 32 + kb], agp + k1);
      async_g2l_b128(&Bs[buf ^ 1][row * 32 + kb], bgp + k1);
    }
#pragma unroll
    for (int i = 0; i < 2; ++i) {
      v16h a = frag_a(&As[buf][0] + tm[i] * 16 * 32, 32);
      v16h b = frag_bt(&Bs[buf][0] + tn[i] * 16 * 32, 32);
      acc[i] = wmma16(a, b, acc[i]);
    }
    wait_async0();
    __syncthreads();
  }

  int n_, mb; cpos(n_, mb);
#pragma unroll
  for (int i = 0; i < 2; ++i) {
#pragma unroll
    for (int j = 0; j < 8; ++j) {
      int mm = m0 + tm[i] * 16 + mb + j;
      int nn = n0 + tn[i] * 16 + n_;
      out[(size_t)mm * DD + nn] = (_Float16)(acc[i][j] + bias[nn]);
    }
  }
}

// -------------------------- per-(b,h) attention -----------------------------
// grid (32, 8, 7): batch, head, 32-row chunk. S=QK^T/8, softmax, P@V.
// K tile (196x64, row stride 512) is DMA'd by the Tensor Data Mover; rows
// 196..207 of the padded tile stay uninitialized but only pollute S columns
// m>=196, which the softmax masks and overwrites with exact zeros.
__global__ __launch_bounds__(256) void attn_kernel(
    const _Float16* qh, const _Float16* kh, const _Float16* vh, float* ctx) {
  __shared__ __align__(16) float S[32 * 224];
  __shared__ __align__(16) _Float16 U[208 * 64 + 32 * 64];  // K|Q then V^T
  _Float16* Ks = U;                 // [208][64]
  _Float16* Qs = U + 208 * 64;      // [32][64]
  _Float16* Vt = U;                 // [64][224] (phase 2, overlays K|Q)

  const int b = blockIdx.x, h = blockIdx.y, ch = blockIdx.z;
  const int tid = threadIdx.x, wave = tid >> 5, lane = tid & 31;
  const int l0 = ch * 32;

#if HAVE_TDM
  if (wave == 0) {
    tdm_load_2d_f16(Ks, kh + (size_t)(b * LL) * DD + h * DHD,
                    /*tileW=*/DHD, /*tileH=*/LL, /*rowStride=*/DD,
                    /*tensorW=*/DD, /*tensorH=*/MM);
    wait_tensor0();
  }
#else
#pragma unroll
  for (int i = 0; i < 7; ++i) {
    int e = tid + 256 * i;
    if (e < 1664) {
      int krow = e >> 3, kkb = (e & 7) * 8;
      int m = krow > LL - 1 ? LL - 1 : krow;
      async_g2l_b128(Ks + krow * 64 + kkb,
                     kh + ((size_t)(b * LL + m)) * DD + h * DHD + kkb);
    }
  }
#endif
  {  // stage Q chunk (32x64, clamped rows) via async copies
    int row = tid >> 2, kb = (tid & 3) * 8;
    int l = l0 + row; if (l > LL - 1) l = LL - 1;
    async_g2l_b128(Qs + row * 64 + kb,
                   qh + ((size_t)(b * LL + l)) * DD + h * DHD + kb);
  }
  wait_async0();
  __syncthreads();

  // S tiles: 2 (M) x 13 (N over 208 K-rows); t = wave + 8*i
  int n_, mb; cpos(n_, mb);
#pragma unroll
  for (int i = 0; i < 4; ++i) {
    int t = wave + 8 * i;
    if (t < 26) {
      int mt = t / 13, nt = t - mt * 13;
      v8f acc = vzero();
#pragma unroll
      for (int ks = 0; ks < 2; ++ks) {
        v16h a = frag_a(Qs + mt * 16 * 64 + ks * 32, 64);
        v16h bfr = frag_bt(Ks + nt * 16 * 64 + ks * 32, 64);
        acc = wmma16(a, bfr, acc);
      }
#pragma unroll
      for (int j = 0; j < 8; ++j)
        S[(mt * 16 + mb + j) * 224 + nt * 16 + n_] = 0.125f * acc[j];
    }
  }
  __syncthreads();

  // softmax: wave handles 4 rows; lane strides columns by 32 (224 total).
#pragma unroll
  for (int jj = 0; jj < 4; ++jj) {
    int r = wave * 4 + jj;
    float pv[7];
    float mx = -3.4e38f;
#pragma unroll
    for (int ii = 0; ii < 7; ++ii) {
      int m = lane + 32 * ii;
      float s = (m < LL) ? S[r * 224 + m] : -3.4e38f;
      pv[ii] = s;
      mx = fmaxf(mx, s);
    }
#pragma unroll
    for (int off = 16; off > 0; off >>= 1) mx = fmaxf(mx, __shfl_xor(mx, off));
    float sum = 0.f;
#pragma unroll
    for (int ii = 0; ii < 7; ++ii) {
      int m = lane + 32 * ii;
      float p = (m < LL) ? __expf(pv[ii] - mx) : 0.f;
      pv[ii] = p;
      sum += p;
    }
#pragma unroll
    for (int off = 16; off > 0; off >>= 1) sum += __shfl_xor(sum, off);
    float inv = 1.0f / sum;
#pragma unroll
    for (int ii = 0; ii < 7; ++ii) {
      int m = lane + 32 * ii;
      S[r * 224 + m] = pv[ii] * inv;   // cols >= 196 become exact zeros
    }
  }
  __syncthreads();

  // stage V transposed [d][m] with zero padding for m in [196,224)
  if (tid < 224) {
    int m = tid;
    if (m < LL) {
      const _Float16* src = vh + ((size_t)(b * LL + m)) * DD + h * DHD;
#pragma unroll
      for (int d = 0; d < 64; ++d) Vt[d * 224 + m] = src[d];
    } else {
#pragma unroll
      for (int d = 0; d < 64; ++d) Vt[d * 224 + m] = (_Float16)0.f;
    }
  }
  __syncthreads();

  // ctx = P @ V : 2 (M) x 4 (N=64/16) tiles, one per wave; K=224 (7 steps)
  {
    int mt = wave >> 2, nt = wave & 3;
    v8f acc = vzero();
#pragma unroll
    for (int ks = 0; ks < 7; ++ks) {
      v16h a = frag_a_f32(S + mt * 16 * 224 + ks * 32, 224);
      v16h bfr = frag_bt(Vt + nt * 16 * 224 + ks * 32, 224);
      acc = wmma16(a, bfr, acc);
    }
#pragma unroll
    for (int j = 0; j < 8; ++j) {
      int l = l0 + mt * 16 + mb + j;
      if (l < LL)
        ctx[((size_t)(b * LL + l)) * DD + h * DHD + nt * 16 + n_] = acc[j];
    }
  }
}

// --------------------- residual + LayerNorm (f32, in place) ----------------
__global__ __launch_bounds__(256) void ln_kernel(
    const float* ctxb, float* x, _Float16* xh, const float* g,
    const float* beta) {
  __shared__ float red[8];
  __shared__ float bc0, bc1;
  const int m = blockIdx.x, tid = threadIdx.x;
  const int wave = tid >> 5, lane = tid & 31;
  const float* crow = ctxb + (size_t)m * DD;
  float* xrow = x + (size_t)m * DD;
  float v0 = crow[tid] + xrow[tid];
  float v1 = crow[tid + 256] + xrow[tid + 256];
  float s = v0 + v1;
#pragma unroll
  for (int off = 16; off > 0; off >>= 1) s += __shfl_xor(s, off);
  if (lane == 0) red[wave] = s;
  __syncthreads();
  if (tid == 0) {
    float t = 0;
#pragma unroll
    for (int i = 0; i < 8; ++i) t += red[i];
    bc0 = t * (1.0f / DD);
  }
  __syncthreads();
  const float mean = bc0;
  float d0 = v0 - mean, d1 = v1 - mean;
  float s2 = d0 * d0 + d1 * d1;
#pragma unroll
  for (int off = 16; off > 0; off >>= 1) s2 += __shfl_xor(s2, off);
  if (lane == 0) red[wave] = s2;
  __syncthreads();
  if (tid == 0) {
    float t = 0;
#pragma unroll
    for (int i = 0; i < 8; ++i) t += red[i];
    bc1 = rsqrtf(t * (1.0f / DD) + 1e-6f);
  }
  __syncthreads();
  const float inv = bc1;
  float o0 = d0 * inv * g[tid] + beta[tid];
  float o1 = d1 * inv * g[tid + 256] + beta[tid + 256];
  xrow[tid] = o0; xrow[tid + 256] = o1;
  xh[(size_t)m * DD + tid] = (_Float16)o0;
  xh[(size_t)m * DD + tid + 256] = (_Float16)o1;
}

// ------------- cross-batch retrieval: r tile + max reductions --------------
// grid 1024 = (a,b). Full 196x196 (padded 208) tile, K=512; no r storage.
// Async double-buffered k-slice staging.
__global__ __launch_bounds__(256) void retrieve_kernel(
    const _Float16* qh, const _Float16* kh, const float* lsp, float* cw) {
  __shared__ __align__(16) _Float16 As[2][208 * 32];
  __shared__ __align__(16) _Float16 Bs[2][208 * 32];
  __shared__ unsigned rowmax[208];
  __shared__ unsigned colmax[208];
  __shared__ float partial[8];
  const int a = blockIdx.x >> 5, b = blockIdx.x & 31;
  const int tid = threadIdx.x, wave = tid >> 5, lane = tid & 31;
  for (int i = tid; i < 208; i += 256) { rowmax[i] = 0u; colmax[i] = 0u; }

  v8f acc[22];
#pragma unroll
  for (int i = 0; i < 22; ++i) acc[i] = vzero();
  const _Float16* Ag = qh + (size_t)a * LL * DD;
  const _Float16* Bg = kh + (size_t)b * LL * DD;

#pragma unroll
  for (int i = 0; i < 4; ++i) {
    int e = tid + 256 * i;
    if (e < 832) {
      int row = e >> 2, kbx = (e & 3) * 8;
      int rc = row > LL - 1 ? LL - 1 : row;
      async_g2l_b128(&As[0][row * 32 + kbx], Ag + (size_t)rc * DD + kbx);
      async_g2l_b128(&Bs[0][row * 32 + kbx], Bg + (size_t)rc * DD + kbx);
    }
  }
  wait_async0();
  __syncthreads();

  int buf = 0;
  for (int k0 = 0; k0 < DD; k0 += 32, buf ^= 1) {
    const int k1 = k0 + 32;
    if (k1 < DD) {
#pragma unroll
      for (int i = 0; i < 4; ++i) {
        int e = tid + 256 * i;
        if (e < 832) {
          int row = e >> 2, kbx = (e & 3) * 8;
          int rc = row > LL - 1 ? LL - 1 : row;
          async_g2l_b128(&As[buf ^ 1][row * 32 + kbx],
                         Ag + (size_t)rc * DD + k1 + kbx);
          async_g2l_b128(&Bs[buf ^ 1][row * 32 + kbx],
                         Bg + (size_t)rc * DD + k1 + kbx);
        }
      }
    }
#pragma unroll
    for (int i = 0; i < 22; ++i) {
      int t = wave + 8 * i;
      if (t < 169) {
        int mt = t / 13, nt = t - mt * 13;
        v16h af = frag_a(&As[buf][0] + mt * 16 * 32, 32);
        v16h bf = frag_bt(&Bs[buf][0] + nt * 16 * 32, 32);
        acc[i] = wmma16(af, bf, acc[i]);
      }
    }
    wait_async0();
    __syncthreads();
  }

  int n_, mb; cpos(n_, mb);
#pragma unroll
  for (int i = 0; i < 22; ++i) {
    int t = wave + 8 * i;
    if (t < 169) {
      int mt = t / 13, nt = t - mt * 13;
      int mcol = nt * 16 + n_;
      if (mcol < LL) {
        float cmax = -3.4e38f;
#pragma unroll
        for (int j = 0; j < 8; ++j) {
          int l = mt * 16 + mb + j;
          if (l < LL) {
            float v = acc[i][j];
            atomicMax(&rowmax[l], ordkey(v));
            cmax = fmaxf(cmax, v);
          }
        }
        atomicMax(&colmax[mcol], ordkey(cmax));
      }
    }
  }
  __syncthreads();

  float s = 0.f;
  if (tid < LL) s = orddec(rowmax[tid]) + orddec(colmax[tid]);
#pragma unroll
  for (int off = 16; off > 0; off >>= 1) s += __shfl_xor(s, off);
  if (lane == 0) partial[wave] = s;
  __syncthreads();
  if (tid == 0) {
    float tot = 0;
#pragma unroll
    for (int i = 0; i < 8; ++i) tot += partial[i];
    cw[a * NB + b] = __expf(*lsp) * (tot * (1.0f / LL)) * 0.5f;
  }
}

// --------------------------- final FC + ReLU -------------------------------
// grid (98, 8). A = cat(x1h, x2h) [6272, 1024], B^T = fcT [512][1024].
__global__ __launch_bounds__(256) void fc_kernel(
    const _Float16* x1h, const _Float16* x2h, const _Float16* fcT,
    const float* fcb, float* out) {
  __shared__ __align__(16) _Float16 As[2][64 * 32];
  __shared__ __align__(16) _Float16 Bs[2][64 * 32];
  const int tid = threadIdx.x, wave = tid >> 5;
  const int m0 = blockIdx.x * 64, n0 = blockIdx.y * 64;
  int tm[2], tn[2];
  v8f acc[2];
#pragma unroll
  for (int i = 0; i < 2; ++i) {
    int t = wave * 2 + i;
    tm[i] = t >> 2; tn[i] = t & 3;
    acc[i] = vzero();
  }
  const int row = tid >> 2, kb = (tid & 3) * 8;
  const _Float16* bgp = fcT + (size_t)(n0 + row) * CC + kb;

  // A source selects x1h for k<512, x2h for k>=512 (kb chunk never straddles)
  {
    const _Float16* src = x1h + (size_t)(m0 + row) * DD + kb;
    async_g2l_b128(&As[0][row * 32 + kb], src);
    async_g2l_b128(&Bs[0][row * 32 + kb], bgp);
  }
  wait_async0();
  __syncthreads();

  int buf = 0;
  for (int k0 = 0; k0 < CC; k0 += 32, buf ^= 1) {
    const int k1 = k0 + 32;
    if (k1 < CC) {
      int kg = k1 + kb;
      const _Float16* src = (kg < DD)
          ? (x1h + (size_t)(m0 + row) * DD + kg)
          : (x2h + (size_t)(m0 + row) * DD + (kg - DD));
      async_g2l_b128(&As[buf ^ 1][row * 32 + kb], src);
      async_g2l_b128(&Bs[buf ^ 1][row * 32 + kb], bgp + k1);
    }
#pragma unroll
    for (int i = 0; i < 2; ++i) {
      v16h a = frag_a(&As[buf][0] + tm[i] * 16 * 32, 32);
      v16h b = frag_bt(&Bs[buf][0] + tn[i] * 16 * 32, 32);
      acc[i] = wmma16(a, b, acc[i]);
    }
    wait_async0();
    __syncthreads();
  }

  int n_, mb; cpos(n_, mb);
#pragma unroll
  for (int i = 0; i < 2; ++i) {
#pragma unroll
    for (int j = 0; j < 8; ++j) {
      int mm = m0 + tm[i] * 16 + mb + j;
      int nn = n0 + tn[i] * 16 + n_;
      float v = acc[i][j] + fcb[nn];
      out[(size_t)mm * DD + nn] = v > 0.f ? v : 0.f;
    }
  }
}

// ------------------------------- host side ---------------------------------

extern "C" void kernel_launch(void* const* d_in, const int* in_sizes, int n_in,
                              void* d_out, int out_size, void* d_ws,
                              size_t ws_size, hipStream_t stream) {
  (void)in_sizes; (void)n_in; (void)out_size; (void)ws_size;
  const float* in1  = (const float*)d_in[0];
  const float* in2  = (const float*)d_in[1];
  const float* cwv  = (const float*)d_in[2];
  const float* cbv  = (const float*)d_in[3];
  const float* wemb = (const float*)d_in[4];
  const float* hemb = (const float*)d_in[5];
  const float* Wq   = (const float*)d_in[6];
  const float* bq   = (const float*)d_in[7];
  const float* Wk   = (const float*)d_in[8];
  const float* bk   = (const float*)d_in[9];
  const float* Wv   = (const float*)d_in[10];
  const float* bv   = (const float*)d_in[11];
  const float* lng  = (const float*)d_in[12];
  const float* lnb  = (const float*)d_in[13];
  const float* lsc  = (const float*)d_in[14];
  const float* fcw  = (const float*)d_in[15];
  const float* fcb  = (const float*)d_in[16];

  float* out  = (float*)d_out;                 // [32,196,512]
  float* cw1  = out + (size_t)MM * DD;         // [32,32]
  float* cw2  = cw1 + NB * NB;                 // [32,32]
  float* pre1 = cw2 + NB * NB;                 // [32,196,512]
  float* pre2 = pre1 + (size_t)MM * DD;        // [32,196,512]

  char* ws = (char*)d_ws;
  const size_t F32 = (size_t)MM * DD * sizeof(float);      // 12,845,056
  const size_t F16 = (size_t)MM * DD * sizeof(_Float16);   //  6,422,528
  float*    x1f = (float*)(ws);
  float*    x2f = (float*)(ws + F32);
  _Float16* x1h = (_Float16*)(ws + 2 * F32);
  _Float16* x2h = (_Float16*)(ws + 2 * F32 + F16);
  _Float16* qh  = (_Float16*)(ws + 2 * F32 + 2 * F16);
  _Float16* kh  = (_Float16*)(ws + 2 * F32 + 3 * F16);
  _Float16* vh  = (_Float16*)(ws + 2 * F32 + 4 * F16);
  float*    ctx = (float*)(ws + 2 * F32 + 5 * F16);
  _Float16* wT  = (_Float16*)(ws + 3 * F32 + 5 * F16);
  _Float16* fcT = (_Float16*)(ws + 3 * F32 + 5 * F16 +
                              (size_t)2 * 3 * DD * DD * sizeof(_Float16));

  wconv_kernel<<<dim3(8192), dim3(256), 0, stream>>>(Wq, Wk, Wv, fcw, wT, fcT);
  embed_kernel<<<dim3(98, 8, 2), dim3(256), 0, stream>>>(
      in1, in2, cwv, cbv, wemb, hemb, pre1, pre2, x1f, x2f, x1h, x2h);

  for (int layer = 0; layer < 2; ++layer) {
    const _Float16* wTl = wT + (size_t)layer * 3 * DD * DD;
    const float* bql = bq + layer * DD;
    const float* bkl = bk + layer * DD;
    const float* bvl = bv + layer * DD;
    const float* gl = lng + layer * DD;
    const float* bl = lnb + layer * DD;

    // call 1: x1' = attn(q=x1, kv=x2)
    qkv_kernel<<<dim3(98, 8, 3), dim3(256), 0, stream>>>(
        x1h, x2h, wTl, bql, bkl, bvl, qh, kh, vh);
    attn_kernel<<<dim3(32, 8, 7), dim3(256), 0, stream>>>(qh, kh, vh, ctx);
    if (layer == 1)
      retrieve_kernel<<<dim3(1024), dim3(256), 0, stream>>>(qh, kh,
                                                            lsc + layer, cw1);
    // call 2 projections use OLD x1 -> run before LN overwrites x1h
    qkv_kernel<<<dim3(98, 8, 3), dim3(256), 0, stream>>>(
        x2h, x1h, wTl, bql, bkl, bvl, qh, kh, vh);
    ln_kernel<<<dim3(MM), dim3(256), 0, stream>>>(ctx, x1f, x1h, gl, bl);
    attn_kernel<<<dim3(32, 8, 7), dim3(256), 0, stream>>>(qh, kh, vh, ctx);
    if (layer == 1)
      retrieve_kernel<<<dim3(1024), dim3(256), 0, stream>>>(qh, kh,
                                                            lsc + layer, cw2);
    ln_kernel<<<dim3(MM), dim3(256), 0, stream>>>(ctx, x2f, x2h, gl, bl);
  }

  fc_kernel<<<dim3(98, 8), dim3(256), 0, stream>>>(x1h, x2h, fcT, fcb, out);
}
